// FlashLlamaAttention_30700426232363
// MI455X (gfx1250) — compile-verified, matrix operand order checked by reading, not run on hardware
//
#include <hip/hip_runtime.h>
#include <hip/hip_bf16.h>
#include <stdint.h>
#include <stddef.h>

typedef __bf16 bf16_t;
typedef __attribute__((ext_vector_type(16))) __bf16 v16bf;
typedef __attribute__((ext_vector_type(8)))  __bf16 v8bf;
typedef __attribute__((ext_vector_type(8)))  float  v8f;
typedef __attribute__((ext_vector_type(4)))  int    i32x4;

#define B_    2
#define S_    1024
#define HID_  4096
#define NH_   32
#define NKV_  8
#define D_    128
#define QKVN  6144          // (NH + 2*NKV) * D
#define SCALE_ 0.08838834764831845f  // 128^-0.5

// ---- feature probes (compile-safe: fall back if builtins are absent) ----
#if defined(__has_builtin)
#if __has_builtin(__builtin_amdgcn_global_load_async_to_lds_b128)
#define HAVE_ASYNC_LDS 1
#endif
#if __has_builtin(__builtin_amdgcn_s_wait_asynccnt)
#define HAVE_WAIT_ASYNC_BUILTIN 1
#endif
#endif

#if defined(HAVE_ASYNC_LDS)
// Param types per hipcc diagnostic: pointer to int __vector(4) with
// global (__device__/AS1) and shared (__shared__/AS3) address spaces.
typedef __attribute__((address_space(1))) i32x4* gv4_t;
typedef __attribute__((address_space(3))) i32x4* lv4_t;
__device__ __forceinline__ void wait_async0() {
#if defined(HAVE_WAIT_ASYNC_BUILTIN)
  __builtin_amdgcn_s_wait_asynccnt(0);
#else
  asm volatile("s_wait_asynccnt 0x0" ::: "memory");
#endif
}
__device__ __forceinline__ void async_copy16(const void* g, void* l) {
  __builtin_amdgcn_global_load_async_to_lds_b128(
      (gv4_t)(uintptr_t)g, (lv4_t)(uintptr_t)l, 0, 0);
}
#endif

// ---- fp32 -> bf16 (round to nearest even), via bit ops
__device__ __forceinline__ bf16_t f2bf(float f) {
  union { float f; uint32_t u; } x; x.f = f;
  uint32_t r = x.u + 0x7FFFu + ((x.u >> 16) & 1u);
  uint16_t h = (uint16_t)(r >> 16);
  bf16_t o; __builtin_memcpy(&o, &h, 2); return o;
}

__device__ __forceinline__ v16bf pack16(v8bf lo, v8bf hi) {
  v16bf r;
#pragma unroll
  for (int e = 0; e < 8; e++) { r[e] = lo[e]; r[e + 8] = hi[e]; }
  return r;
}

// ---------------------------------------------------------------- converts
__global__ void cvt_f32_bf16(const float* __restrict__ in, bf16_t* __restrict__ out, size_t n) {
  size_t i = (size_t)blockIdx.x * blockDim.x + threadIdx.x;
  size_t stride = (size_t)gridDim.x * blockDim.x;
  for (; i < n; i += stride) out[i] = f2bf(in[i]);
}

// in[K][N] f32 -> out[N][K] bf16, LDS-tiled 32x32 transpose
__global__ __launch_bounds__(256) void transpose_cvt(
    const float* __restrict__ in, bf16_t* __restrict__ out, int K, int N) {
  __shared__ float tile[32][33];
  int k0 = blockIdx.y * 32;
  int n0 = blockIdx.x * 32;
  int tx = threadIdx.x & 31;
  int ty = threadIdx.x >> 5;   // 0..7
#pragma unroll
  for (int j = 0; j < 4; j++)
    tile[ty + j * 8][tx] = in[(size_t)(k0 + ty + j * 8) * N + n0 + tx];
  __syncthreads();
#pragma unroll
  for (int j = 0; j < 4; j++)
    out[(size_t)(n0 + ty + j * 8) * K + k0 + tx] = f2bf(tile[tx][ty + j * 8]);
}

// ---------------------------------------------------------------- GEMM
// C[M,N](f32) = A[M,K](bf16 row-major) * B[K,N], with B given TRANSPOSED:
// BT[N][K] bf16 row-major.  Block: 256 thr = 8 waves; tile M=128, N=128; K step 32.
// BT k-slice (128 rows x 32 k; LDS row stride 40 elems) double-buffered in LDS:
// the async copy of tile i+1 overlaps the 8 WMMAs of tile i.
__global__ __launch_bounds__(256) void gemm_bf16_wmma(
    const bf16_t* __restrict__ A, const bf16_t* __restrict__ BT,
    float* __restrict__ C, int M, int N, int Kd) {
  __shared__ __align__(16) bf16_t bt[2][128 * 40];
  const int lane = threadIdx.x & 31;
  const int wave = threadIdx.x >> 5;
  const int lm   = lane & 15;
  const int half = lane >> 4;
  const int n0 = blockIdx.x * 128;
  const int m0 = blockIdx.y * 128 + wave * 16;

  // staging map: thread t copies 16 elems of row sn (two 16B chunks)
  const int sn  = threadIdx.x >> 1;          // 0..127
  const int sk0 = (threadIdx.x & 1) * 16;    // 0 or 16
  const bf16_t* gsrc = BT + (size_t)(n0 + sn) * Kd + sk0;
  bf16_t* lds0 = &bt[0][sn * 40 + sk0];
  bf16_t* lds1 = &bt[1][sn * 40 + sk0];

  v8f acc[8];
#pragma unroll
  for (int t = 0; t < 8; t++)
#pragma unroll
    for (int e = 0; e < 8; e++) acc[t][e] = 0.f;

  const bf16_t* arow = A + (size_t)(m0 + lm) * Kd;

  // prologue: stage tile 0 into buffer 0
#if defined(HAVE_ASYNC_LDS)
  async_copy16(gsrc, lds0);
  async_copy16(gsrc + 8, lds0 + 8);
#else
  *(v8bf*)lds0       = *(const v8bf*)(gsrc);
  *(v8bf*)(lds0 + 8) = *(const v8bf*)(gsrc + 8);
#endif

  int p = 0;
  for (int kc = 0; kc < Kd; kc += 32, p ^= 1) {
#if defined(HAVE_ASYNC_LDS)
    wait_async0();              // this wave's slice of tile kc has landed
#endif
    __syncthreads();            // => every wave's slice has landed

    // stage tile kc+32 into the other buffer; overlaps with compute below
    if (kc + 32 < Kd) {
      bf16_t* dst = p ? lds0 : lds1;
#if defined(HAVE_ASYNC_LDS)
      async_copy16(gsrc + kc + 32, dst);
      async_copy16(gsrc + kc + 40, dst + 8);
#else
      *(v8bf*)dst       = *(const v8bf*)(gsrc + kc + 32);
      *(v8bf*)(dst + 8) = *(const v8bf*)(gsrc + kc + 40);
#endif
    }

    v16bf af = pack16(*(const v8bf*)(arow + kc + half * 8),
                      *(const v8bf*)(arow + kc + 16 + half * 8));
#pragma unroll
    for (int nt = 0; nt < 8; nt++) {
      const bf16_t* bp = &bt[p][(nt * 16 + lm) * 40];
      v16bf bfv = pack16(*(const v8bf*)(bp + half * 8),
                         *(const v8bf*)(bp + 16 + half * 8));
      acc[nt] = __builtin_amdgcn_wmma_f32_16x16x32_bf16(
          false, af, false, bfv, (short)0, acc[nt], false, false);
    }
  }
#pragma unroll
  for (int nt = 0; nt < 8; nt++)
#pragma unroll
    for (int v = 0; v < 8; v++)
      C[(size_t)(m0 + v + half * 8) * N + n0 + nt * 16 + lm] = acc[nt][v];
}

// ---------------------------------------------------------------- RoPE + head split
__global__ void rope_split(const float* __restrict__ qkv, const float* __restrict__ cosT,
                           const float* __restrict__ sinT, bf16_t* __restrict__ Qo,
                           bf16_t* __restrict__ Ko) {
  int idx = blockIdx.x * 256 + threadIdx.x;   // B*S*40*64 exactly
  int i  = idx & 63;
  int t  = idx >> 6;
  int hh = t % (NH_ + NKV_);
  t /= (NH_ + NKV_);
  int s = t % S_;
  int b = t / S_;
  bool isQ = hh < NH_;
  int h = isQ ? hh : hh - NH_;
  const float* base = qkv + (size_t)(b * S_ + s) * QKVN + (isQ ? h * D_ : HID_ + h * D_);
  float x1 = base[2 * i], x2 = base[2 * i + 1];
  float c = cosT[s * 64 + i], sn = sinT[s * 64 + i];
  bf16_t* o = isQ ? Qo + ((size_t)(b * NH_ + h) * S_ + s) * D_
                  : Ko + ((size_t)(b * NKV_ + h) * S_ + s) * D_;
  o[2 * i]     = f2bf(x1 * c - x2 * sn);
  o[2 * i + 1] = f2bf(x1 * sn + x2 * c);
}

// qkv f32 -> V^T bf16 [B,NKV,D,S]
__global__ void v_transpose(const float* __restrict__ qkv, bf16_t* __restrict__ VT) {
  int idx = blockIdx.x * 256 + threadIdx.x;   // B*NKV*D*S exactly
  int s  = idx & (S_ - 1);
  int d  = (idx >> 10) & (D_ - 1);
  int kv = (idx >> 17) & (NKV_ - 1);
  int b  = idx >> 20;
  VT[idx] = f2bf(qkv[(size_t)(b * S_ + s) * QKVN + (NH_ + NKV_) * D_ + kv * D_ + d]);
}

// ---------------------------------------------------------------- flash attention
// One wave = one (b, head, 16-row q tile). Keys in blocks of 32.
__global__ __launch_bounds__(256) void flash_attn(
    const bf16_t* __restrict__ Q, const bf16_t* __restrict__ K,
    const bf16_t* __restrict__ VT, bf16_t* __restrict__ Oo) {
  __shared__ __align__(16) bf16_t pbuf[8][16 * 32];
  const int lane = threadIdx.x & 31;
  const int wave = threadIdx.x >> 5;
  const int lm   = lane & 15;
  const int half = lane >> 4;
  int task = blockIdx.x * 8 + wave;          // B*NH*(S/16) = 4096 tasks
  int qt = task & 63;
  int h  = (task >> 6) & 31;
  int b  = task >> 11;
  int q0 = qt * 16;
  int kvh = h >> 2;                          // GQA: G=4

  const bf16_t* qbase = Q + ((size_t)(b * NH_ + h) * S_ + q0 + lm) * D_;
  v16bf qa[4];
#pragma unroll
  for (int c = 0; c < 4; c++)
    qa[c] = pack16(*(const v8bf*)(qbase + c * 32 + half * 8),
                   *(const v8bf*)(qbase + c * 32 + 16 + half * 8));

  v8f o[8];
#pragma unroll
  for (int t = 0; t < 8; t++)
#pragma unroll
    for (int e = 0; e < 8; e++) o[t][e] = 0.f;
  float mrow[8], lrow[8];
#pragma unroll
  for (int v = 0; v < 8; v++) { mrow[v] = -3.0e38f; lrow[v] = 0.f; }

  const bf16_t* kbase = K + (size_t)(b * NKV_ + kvh) * S_ * D_;
  const bf16_t* vbase = VT + (size_t)(b * NKV_ + kvh) * D_ * S_;

  for (int kb = 0; kb < q0 + 16; kb += 32) {
    v8f sA, sB;
#pragma unroll
    for (int e = 0; e < 8; e++) { sA[e] = 0.f; sB[e] = 0.f; }
#pragma unroll
    for (int c = 0; c < 4; c++) {
      const bf16_t* k0p = kbase + (size_t)(kb + lm) * D_ + c * 32;
      v16bf bA = pack16(*(const v8bf*)(k0p + half * 8),
                        *(const v8bf*)(k0p + 16 + half * 8));
      sA = __builtin_amdgcn_wmma_f32_16x16x32_bf16(false, qa[c], false, bA, (short)0, sA, false, false);
      const bf16_t* k1p = kbase + (size_t)(kb + 16 + lm) * D_ + c * 32;
      v16bf bB = pack16(*(const v8bf*)(k1p + half * 8),
                        *(const v8bf*)(k1p + 16 + half * 8));
      sB = __builtin_amdgcn_wmma_f32_16x16x32_bf16(false, qa[c], false, bB, (short)0, sB, false, false);
    }
    float rmax[8];
#pragma unroll
    for (int v = 0; v < 8; v++) {
      int row = q0 + v + half * 8;
      float a = sA[v] * SCALE_;  if (kb + lm > row)      a = -3.0e38f;
      float bb = sB[v] * SCALE_; if (kb + 16 + lm > row) bb = -3.0e38f;
      sA[v] = a; sB[v] = bb;
      rmax[v] = fmaxf(a, bb);
    }
#pragma unroll
    for (int off = 1; off < 16; off <<= 1)
#pragma unroll
      for (int v = 0; v < 8; v++) rmax[v] = fmaxf(rmax[v], __shfl_xor(rmax[v], off, 32));

    float alpha[8], rsum[8];
#pragma unroll
    for (int v = 0; v < 8; v++) {
      float mn = fmaxf(mrow[v], rmax[v]);
      alpha[v] = __expf(mrow[v] - mn);
      mrow[v] = mn;
      float pa = __expf(sA[v] - mn);
      float pb = __expf(sB[v] - mn);
      pbuf[wave][(v + half * 8) * 32 + lm]      = f2bf(pa);
      pbuf[wave][(v + half * 8) * 32 + 16 + lm] = f2bf(pb);
      rsum[v] = pa + pb;
    }
#pragma unroll
    for (int off = 1; off < 16; off <<= 1)
#pragma unroll
      for (int v = 0; v < 8; v++) rsum[v] += __shfl_xor(rsum[v], off, 32);
#pragma unroll
    for (int v = 0; v < 8; v++) lrow[v] = lrow[v] * alpha[v] + rsum[v];

    const bf16_t* pr = &pbuf[wave][lm * 32];
    v16bf pf = pack16(*(const v8bf*)(pr + half * 8),
                      *(const v8bf*)(pr + 16 + half * 8));
#pragma unroll
    for (int dt = 0; dt < 8; dt++) {
#pragma unroll
      for (int v = 0; v < 8; v++) o[dt][v] *= alpha[v];
      const bf16_t* vp = vbase + (size_t)(dt * 16 + lm) * S_ + kb;
      v16bf vf = pack16(*(const v8bf*)(vp + half * 8),
                        *(const v8bf*)(vp + 16 + half * 8));
      o[dt] = __builtin_amdgcn_wmma_f32_16x16x32_bf16(false, pf, false, vf, (short)0, o[dt], false, false);
    }
  }

  float inv[8];
#pragma unroll
  for (int v = 0; v < 8; v++) inv[v] = 1.0f / lrow[v];
#pragma unroll
  for (int dt = 0; dt < 8; dt++)
#pragma unroll
    for (int v = 0; v < 8; v++)
      Oo[((size_t)b * S_ + q0 + v + half * 8) * HID_ + h * D_ + dt * 16 + lm] =
          f2bf(o[dt][v] * inv[v]);
}

// ---------------------------------------------------------------- launch
extern "C" void kernel_launch(void* const* d_in, const int* in_sizes, int n_in,
                              void* d_out, int out_size, void* d_ws, size_t ws_size,
                              hipStream_t stream) {
  (void)in_sizes; (void)n_in; (void)out_size; (void)ws_size;
  const float* hidden = (const float*)d_in[0];
  const float* cosT   = (const float*)d_in[1];
  const float* sinT   = (const float*)d_in[2];
  const float* wqkv   = (const float*)d_in[3];
  const float* wo     = (const float*)d_in[4];
  float* out = (float*)d_out;

  char* ws = (char*)d_ws;
  size_t off = 0;
  auto alloc = [&](size_t bytes) -> char* {
    char* p = ws + off;
    off += (bytes + 255) & ~(size_t)255;
    return p;
  };
  // region 0: hiddenB (later reused as attention output)
  bf16_t* hiddenB = (bf16_t*)alloc((size_t)B_ * S_ * HID_ * 2);       // 16 MB
  // region 1: wqkvT (later reused for Q / K / V^T)
  char*   reg1    = alloc((size_t)HID_ * QKVN * 2);                   // 48 MB
  bf16_t* woT     = (bf16_t*)alloc((size_t)HID_ * HID_ * 2);          // 32 MB (persistent)
  float*  qkv     = (float*)alloc((size_t)B_ * S_ * QKVN * 4);        // 48 MB

  bf16_t* wqkvT = (bf16_t*)reg1;                                      // [QKVN][HID]
  bf16_t* Qb    = (bf16_t*)reg1;                                      // 16 MB
  bf16_t* Kb    = (bf16_t*)(reg1 + (size_t)B_ * NH_ * S_ * D_ * 2);   //  4 MB
  bf16_t* VTb   = (bf16_t*)(reg1 + (size_t)(B_ * NH_ + B_ * NKV_) * S_ * D_ * 2); // 4 MB
  bf16_t* attnB = hiddenB;

  cvt_f32_bf16<<<2048, 256, 0, stream>>>(hidden, hiddenB, (size_t)B_ * S_ * HID_);
  transpose_cvt<<<dim3(QKVN / 32, HID_ / 32), 256, 0, stream>>>(wqkv, wqkvT, HID_, QKVN);
  transpose_cvt<<<dim3(HID_ / 32, HID_ / 32), 256, 0, stream>>>(wo, woT, HID_, HID_);

  gemm_bf16_wmma<<<dim3(QKVN / 128, (B_ * S_) / 128), 256, 0, stream>>>(
      hiddenB, wqkvT, qkv, B_ * S_, QKVN, HID_);

  rope_split<<<(B_ * S_ * (NH_ + NKV_) * 64) / 256, 256, 0, stream>>>(
      qkv, cosT, sinT, Qb, Kb);
  v_transpose<<<(B_ * NKV_ * D_ * S_) / 256, 256, 0, stream>>>(qkv, VTb);

  flash_attn<<<(B_ * NH_ * (S_ / 16)) / 8, 256, 0, stream>>>(Qb, Kb, VTb, attnB);

  gemm_bf16_wmma<<<dim3(HID_ / 128, (B_ * S_) / 128), 256, 0, stream>>>(
      attnB, woT, out, B_ * S_, HID_, HID_);
}
